// MultiHeadAttention_34162169872901
// MI455X (gfx1250) — matrix operand order repaired and checked by reading, not compiled, not run on hardware
//
#include <hip/hip_runtime.h>
#include <hip/hip_bf16.h>

// ---------------------------------------------------------------------------
// CDNA5 (gfx1250) wave32 WMMA multi-head attention with relative position bias
//  k0: f32->f16 weight pre-convert
//  k1: fused QKV projection (WMMA, f16 weights, f32 activations converted inline)
//  k2: flash attention, K/V tiles staged to LDS via async global->LDS loads
//  k3: output projection (WMMA) + bias, fp32 out
// ---------------------------------------------------------------------------

typedef __attribute__((ext_vector_type(16))) _Float16 v16h;
typedef __attribute__((ext_vector_type(8)))  _Float16 v8h;
typedef __attribute__((ext_vector_type(4)))  _Float16 v4h;
typedef __attribute__((ext_vector_type(8)))  float    v8f;
typedef __attribute__((ext_vector_type(4)))  int      v4i;

#define S_LEN   2048
#define DMODEL  512
#define NHEAD   8
#define DK      64
#define NBIAS   511   // 2*256-1

#if defined(__AMDGCN__) && __has_builtin(__builtin_amdgcn_global_load_async_to_lds_b128)
#define HAVE_ASYNC_LDS 1
#else
#define HAVE_ASYNC_LDS 0
#endif

#if defined(__AMDGCN__) && __has_builtin(__builtin_amdgcn_s_wait_asynccnt)
#define WAIT_ASYNC(n) __builtin_amdgcn_s_wait_asynccnt(n)
#else
#define WAIT_ASYNC(n) asm volatile("s_wait_asynccnt " #n ::: "memory")
#endif

// D = A(16x32 f16) * B(32x16 f16) + C(16x16 f32)
__device__ __forceinline__ v8f wmma16(v16h a, v16h b, v8f c) {
    return __builtin_amdgcn_wmma_f32_16x16x32_f16(
        /*neg_a=*/false, a, /*neg_b=*/false, b,
        /*c_mod=*/(short)0, c, /*reuse_a=*/false, /*reuse_b=*/false);
}

// A-fragment (16x32, M x K), lane holds row (lane&15).
// K halves per ISA: j=0..7 -> K = k0 + 8h + j ; j=8..15 -> K = k0 + 16 + 8h + (j-8)
__device__ __forceinline__ v16h load_a_f32(const float* row, int k0, int lane) {
    int h = (lane >> 4) & 1;
    const float* p0 = row + k0 + 8 * h;
    const float* p1 = row + k0 + 16 + 8 * h;
    v16h a;
#pragma unroll
    for (int j = 0; j < 8; ++j) { a[j] = (_Float16)p0[j]; a[j + 8] = (_Float16)p1[j]; }
    return a;
}

__device__ __forceinline__ v16h load_a_f16(const _Float16* row, int k0, int lane) {
    int h = (lane >> 4) & 1;
    v8h lo = *(const v8h*)(row + k0 + 8 * h);
    v8h hi = *(const v8h*)(row + k0 + 16 + 8 * h);
    v16h a;
#pragma unroll
    for (int j = 0; j < 8; ++j) { a[j] = lo[j]; a[j + 8] = hi[j]; }
    return a;
}

// B-fragment (32x16, K x N), lane holds column (lane&15).
// j=0..7 -> K = k0 + 8h + j ; j=8..15 -> K = k0 + 16 + 8h + (j-8).  colrow = memory row holding that column.
__device__ __forceinline__ v16h load_b_f16(const _Float16* colrow, int k0, int lane) {
    int h = (lane >> 4) & 1;
    const _Float16* p = colrow + k0 + 8 * h;
    v8h lo = *(const v8h*)(p);
    v8h hi = *(const v8h*)(p + 16);
    v16h b;
#pragma unroll
    for (int j = 0; j < 8; ++j) { b[j] = lo[j]; b[j + 8] = hi[j]; }
    return b;
}

// Async (or fallback) 16-byte global -> LDS copy
__device__ __forceinline__ void g2l_b128(const _Float16* g, _Float16* l) {
#if HAVE_ASYNC_LDS
    __builtin_amdgcn_global_load_async_to_lds_b128(
        (__attribute__((address_space(1))) v4i*)g,
        (__attribute__((address_space(3))) v4i*)l, 0, 0);
#else
    *(v8h*)l = *(const v8h*)g;   // global_load_b128 + ds_store_b128
#endif
}

// ---------------------------------------------------------------------------
// Kernel 0: weight f32 -> f16 pre-convert (Wq|Wk|Wv|Wo concatenated)
// ---------------------------------------------------------------------------
__global__ __launch_bounds__(256) void cvt_w_kernel(const float* __restrict__ src,
                                                    _Float16* __restrict__ dst, int n4) {
    int i = blockIdx.x * blockDim.x + threadIdx.x;
    if (i < n4) {
        const float4 f = ((const float4*)src)[i];
        v4h o;
        o[0] = (_Float16)f.x; o[1] = (_Float16)f.y;
        o[2] = (_Float16)f.z; o[3] = (_Float16)f.w;
        ((v4h*)dst)[i] = o;
    }
}

// ---------------------------------------------------------------------------
// Kernel 1: fused Q/K/V projections.  out = X @ W^T  (both K-contiguous).
// grid = (BS/128, DMODEL/64, 3), block = 256 (8 waves); wave tile = 16x64.
// Q,K stored [B][H][S][DK] f16 (Q pre-scaled by 1/8); V stored [B][H][DK][S].
// ---------------------------------------------------------------------------
__global__ __launch_bounds__(256) void qkv_proj_kernel(
    const float* __restrict__ query, const float* __restrict__ key,
    const float* __restrict__ value, const _Float16* __restrict__ Wf,
    _Float16* __restrict__ Qf, _Float16* __restrict__ Kf,
    _Float16* __restrict__ Vt)
{
    const int wave = threadIdx.x >> 5, lane = threadIdx.x & 31;
    const int n = lane & 15, g = (lane >> 4) & 1;
    const int mat = blockIdx.z;
    const float* X = (mat == 0) ? query : (mat == 1) ? key : value;
    const _Float16* W = Wf + (size_t)mat * DMODEL * DMODEL;

    const int m_base = blockIdx.x * 128 + wave * 16;
    const int n_base = blockIdx.y * 64;

    v8f acc[4] = {};
    const float* xrow = X + (size_t)(m_base + n) * DMODEL;
#pragma unroll 4
    for (int k0 = 0; k0 < DMODEL; k0 += 32) {
        if (k0 + 64 < DMODEL) __builtin_prefetch(xrow + k0 + 64, 0, 0);
        v16h a = load_a_f32(xrow, k0, lane);
#pragma unroll
        for (int t = 0; t < 4; ++t) {
            const _Float16* wrow = W + (size_t)(n_base + t * 16 + n) * DMODEL;
            acc[t] = wmma16(a, load_b_f16(wrow, k0, lane), acc[t]);
        }
    }

    const float qscale = (mat == 0) ? 0.125f : 1.0f;  // fold 1/sqrt(dk) into Q
#pragma unroll
    for (int t = 0; t < 4; ++t) {
        const int col = n_base + t * 16 + n;
        const int h = col >> 6, d = col & 63;
#pragma unroll
        for (int v = 0; v < 8; ++v) {
            const int row = m_base + g * 8 + v;
            const int b = row >> 11, s = row & (S_LEN - 1);
            const float val = acc[t][v] * qscale;
            if (mat == 2) {
                Vt[(((size_t)b * NHEAD + h) * DK + d) * S_LEN + s] = (_Float16)val;
            } else {
                const size_t idx = (((size_t)b * NHEAD + h) * S_LEN + s) * DK + d;
                if (mat == 0) Qf[idx] = (_Float16)val;
                else          Kf[idx] = (_Float16)val;
            }
        }
    }
}

// ---------------------------------------------------------------------------
// Kernel 2: flash attention with relative position bias.
// grid = (S/128, B*NHEAD), block = 256 (8 waves); wave owns 16 query rows.
// K/V tiles shared by all 8 waves, double-buffered in LDS, filled with
// async global->LDS loads (ASYNCcnt) when the toolchain exposes them.
// ---------------------------------------------------------------------------
__global__ __launch_bounds__(256) void attn_kernel(
    const _Float16* __restrict__ Qf, const _Float16* __restrict__ Kf,
    const _Float16* __restrict__ Vt, const int* __restrict__ mask,
    const float* __restrict__ rel_bias, _Float16* __restrict__ ctx)
{
    __shared__ __align__(16) _Float16 kbuf[2][32][DK];   // keys x d       (8 KB)
    __shared__ __align__(16) _Float16 vbuf[2][DK][32];   // d    x keys    (8 KB)
    __shared__ __align__(16) _Float16 lds_p[8][16][32];  // per-wave P     (8 KB)
    __shared__ float lds_bias[512];

    const int tid = threadIdx.x;
    const int wave = tid >> 5, lane = tid & 31;
    const int n = lane & 15, g = (lane >> 4) & 1;
    const int bh = blockIdx.y;
    const int b = bh >> 3, h = bh & 7;
    const int qb = blockIdx.x * 128 + wave * 16;

    for (int i = tid; i < NBIAS; i += 256) lds_bias[i] = rel_bias[h * NBIAS + i];

    const size_t head_off = (size_t)bh * S_LEN * DK;
    const _Float16* Qh = Qf + head_off;                 // [S][64]
    const _Float16* Kh = Kf + head_off;                 // [S][64]
    const _Float16* Vh = Vt + (size_t)bh * DK * S_LEN;  // [64][S]
    const int* mb = mask + (size_t)b * S_LEN;

    // Q A-fragments, fixed over the whole key loop
    const _Float16* qrow = Qh + (size_t)(qb + n) * DK;
    const v16h qa0 = load_a_f16(qrow, 0, lane);
    const v16h qa1 = load_a_f16(qrow, 32, lane);

    v8f acc[4] = {};
    float rmax[8], rsum[8];
#pragma unroll
    for (int v = 0; v < 8; ++v) { rmax[v] = -3.0e38f; rsum[v] = 0.0f; }

    // stage one 32-key tile: K 32x64 and V^T 64x32, one b128 per thread each
    auto stage = [&](int kb, int buf) {
        { int r = tid >> 3, c = (tid & 7) * 8;
          g2l_b128(Kh + (size_t)(kb + r) * DK + c, &kbuf[buf][r][c]); }
        { int r = tid >> 2, c = (tid & 3) * 8;
          g2l_b128(Vh + (size_t)r * S_LEN + kb + c, &vbuf[buf][r][c]); }
    };

    stage(0, 0);
    for (int kb = 0, it = 0; kb < S_LEN; kb += 32, ++it) {
        const int buf = it & 1;
        const bool more = (kb + 32) < S_LEN;
        if (more) stage(kb + 32, buf ^ 1);
#if HAVE_ASYNC_LDS
        if (more) { WAIT_ASYNC(2); } else { WAIT_ASYNC(0); }
#endif
        __syncthreads();   // current tile visible to all waves (incl. lds_bias on it 0)

        // ---- scores: S[q, k] = Q . K   (two 16-key halves, B-frags from LDS) ----
        v8f s0 = {}, s1 = {};
        {
            const _Float16* kr0 = &kbuf[buf][n][0];
            s0 = wmma16(qa0, load_b_f16(kr0, 0, lane), s0);
            s0 = wmma16(qa1, load_b_f16(kr0, 32, lane), s0);
            const _Float16* kr1 = &kbuf[buf][16 + n][0];
            s1 = wmma16(qa0, load_b_f16(kr1, 0, lane), s1);
            s1 = wmma16(qa1, load_b_f16(kr1, 32, lane), s1);
        }
        // ---- relative position bias + mask ----
        const float mk0 = mb[kb + n]      ? 0.0f : -1.0e9f;
        const float mk1 = mb[kb + 16 + n] ? 0.0f : -1.0e9f;
#pragma unroll
        for (int v = 0; v < 8; ++v) {
            const int q = qb + g * 8 + v;
            int r0 = (kb + n) - q;      r0 = (r0 < -255) ? -255 : (r0 > 255) ? 255 : r0;
            int r1 = (kb + 16 + n) - q; r1 = (r1 < -255) ? -255 : (r1 > 255) ? 255 : r1;
            s0[v] += lds_bias[r0 + 255] + mk0;
            s1[v] += lds_bias[r1 + 255] + mk1;
        }
        // ---- online softmax (reduce over 16 lanes of each N-group) ----
        float tmax[8], alpha[8], psum[8];
#pragma unroll
        for (int v = 0; v < 8; ++v) tmax[v] = fmaxf(s0[v], s1[v]);
#pragma unroll
        for (int off = 8; off >= 1; off >>= 1) {
#pragma unroll
            for (int v = 0; v < 8; ++v)
                tmax[v] = fmaxf(tmax[v], __shfl_xor(tmax[v], off, 32));
        }
#pragma unroll
        for (int v = 0; v < 8; ++v) {
            const float mnew = fmaxf(rmax[v], tmax[v]);
            alpha[v] = __expf(rmax[v] - mnew);
            rmax[v] = mnew;
            s0[v] = __expf(s0[v] - mnew);
            s1[v] = __expf(s1[v] - mnew);
            psum[v] = s0[v] + s1[v];
        }
#pragma unroll
        for (int off = 8; off >= 1; off >>= 1) {
#pragma unroll
            for (int v = 0; v < 8; ++v)
                psum[v] += __shfl_xor(psum[v], off, 32);
        }
#pragma unroll
        for (int v = 0; v < 8; ++v) {
            rsum[v] = rsum[v] * alpha[v] + psum[v];
            acc[0][v] *= alpha[v]; acc[1][v] *= alpha[v];
            acc[2][v] *= alpha[v]; acc[3][v] *= alpha[v];
        }
        // ---- P: C-layout -> A-layout via per-wave LDS round trip ----
#pragma unroll
        for (int v = 0; v < 8; ++v) {
            lds_p[wave][g * 8 + v][n]      = (_Float16)s0[v];
            lds_p[wave][g * 8 + v][n + 16] = (_Float16)s1[v];
        }
        asm volatile("s_wait_dscnt 0" ::: "memory");
        const v16h pa = load_a_f16(&lds_p[wave][n][0], 0, lane);
        // ---- O += P @ V  (B-frags from LDS V^T tile) ----
#pragma unroll
        for (int dt = 0; dt < 4; ++dt) {
            const _Float16* vcol = &vbuf[buf][dt * 16 + n][0];
            acc[dt] = wmma16(pa, load_b_f16(vcol, 0, lane), acc[dt]);
        }
        __syncthreads();   // all waves done with buf before it is re-filled
    }

    // ---- normalize and write context [B*S][512] f16 ----
    float inv[8];
#pragma unroll
    for (int v = 0; v < 8; ++v)
        inv[v] = (rsum[v] > 0.0f) ? 1.0f / rsum[v] : 0.0f;
#pragma unroll
    for (int v = 0; v < 8; ++v) {
        const int m = qb + g * 8 + v;
        const size_t base = ((size_t)b * S_LEN + m) * DMODEL + (size_t)h * DK;
#pragma unroll
        for (int dt = 0; dt < 4; ++dt)
            ctx[base + dt * 16 + n] = (_Float16)(acc[dt][v] * inv[v]);
    }
}

// ---------------------------------------------------------------------------
// Kernel 3: output projection  out = ctx @ Wo^T + bo  (fp32 out)
// grid = (BS/128, DMODEL/64), block 256; wave tile 16x64.
// ---------------------------------------------------------------------------
__global__ __launch_bounds__(256) void out_proj_kernel(
    const _Float16* __restrict__ ctx, const _Float16* __restrict__ Wof,
    const float* __restrict__ bo, float* __restrict__ out)
{
    const int wave = threadIdx.x >> 5, lane = threadIdx.x & 31;
    const int n = lane & 15, g = (lane >> 4) & 1;
    const int m_base = blockIdx.x * 128 + wave * 16;
    const int n_base = blockIdx.y * 64;

    v8f acc[4] = {};
    const _Float16* crow = ctx + (size_t)(m_base + n) * DMODEL;
#pragma unroll 4
    for (int k0 = 0; k0 < DMODEL; k0 += 32) {
        v16h a = load_a_f16(crow, k0, lane);
#pragma unroll
        for (int t = 0; t < 4; ++t) {
            const _Float16* wrow = Wof + (size_t)(n_base + t * 16 + n) * DMODEL;
            acc[t] = wmma16(a, load_b_f16(wrow, k0, lane), acc[t]);
        }
    }
#pragma unroll
    for (int t = 0; t < 4; ++t) {
        const int col = n_base + t * 16 + n;
        const float bias = bo[col];
#pragma unroll
        for (int v = 0; v < 8; ++v) {
            const int row = m_base + g * 8 + v;
            out[(size_t)row * DMODEL + col] = acc[t][v] + bias;
        }
    }
}

// ---------------------------------------------------------------------------
extern "C" void kernel_launch(void* const* d_in, const int* in_sizes, int n_in,
                              void* d_out, int out_size, void* d_ws, size_t ws_size,
                              hipStream_t stream) {
    (void)in_sizes; (void)n_in; (void)out_size; (void)ws_size;
    const float* query    = (const float*)d_in[0];
    const float* key      = (const float*)d_in[1];
    const float* value    = (const float*)d_in[2];
    const int*   mask     = (const int*)  d_in[3];
    const float* Wq       = (const float*)d_in[4];
    const float* Wk       = (const float*)d_in[5];
    const float* Wv       = (const float*)d_in[6];
    const float* Wo       = (const float*)d_in[7];
    const float* bo       = (const float*)d_in[8];
    const float* rel_bias = (const float*)d_in[9];
    float* out = (float*)d_out;

    const size_t HEAD_ELEMS = (size_t)4 * NHEAD * S_LEN * DK;  // 4,194,304
    const size_t W_ELEMS    = (size_t)DMODEL * DMODEL;         // 262,144
    char* ws = (char*)d_ws;
    _Float16* Qf  = (_Float16*)(ws);
    _Float16* Kf  = (_Float16*)(ws + 1 * HEAD_ELEMS * sizeof(_Float16));
    _Float16* Vt  = (_Float16*)(ws + 2 * HEAD_ELEMS * sizeof(_Float16));
    _Float16* ctx = (_Float16*)(ws + 3 * HEAD_ELEMS * sizeof(_Float16));
    _Float16* Wf  = (_Float16*)(ws + 4 * HEAD_ELEMS * sizeof(_Float16));

    // k0: pre-convert the four weight matrices to f16
    const float* Ws[4] = {Wq, Wk, Wv, Wo};
    for (int i = 0; i < 4; ++i)
        cvt_w_kernel<<<(W_ELEMS / 4 + 255) / 256, 256, 0, stream>>>(
            Ws[i], Wf + (size_t)i * W_ELEMS, (int)(W_ELEMS / 4));

    const int BS = 4 * S_LEN;  // 8192 rows

    dim3 gProj(BS / 128, DMODEL / 64, 3);
    qkv_proj_kernel<<<gProj, 256, 0, stream>>>(query, key, value, Wf, Qf, Kf, Vt);

    dim3 gAttn(S_LEN / 128, 4 * NHEAD);
    attn_kernel<<<gAttn, 256, 0, stream>>>(Qf, Kf, Vt, mask, rel_bias, ctx);

    dim3 gOut(BS / 128, DMODEL / 64);
    out_proj_kernel<<<gOut, 256, 0, stream>>>(ctx, Wf + 3 * W_ELEMS, bo, out);
}
